// Dynamic_Memory_86122684219923
// MI455X (gfx1250) — matrix operand order repaired and checked by reading, not compiled, not run on hardware
//
#include <hip/hip_runtime.h>

#define BH   16      // b*h heads
#define SEQ  4096
#define DH   64
#define NF   256
#define CHK  128
#define NCHK 32
#define NORMF 0.35355339059327379f   // 64^-0.25
#define EPS   1e-6f

typedef __attribute__((ext_vector_type(16))) __bf16    bf16x16;
typedef __attribute__((ext_vector_type(8)))  float     floatx8;
typedef __attribute__((ext_vector_type(8)))  unsigned  uintx8;

union PackU { __bf16 h[2]; unsigned short s[2]; unsigned u; };

__device__ __forceinline__ unsigned short f2bf(float f) {
  PackU p; p.h[0] = (__bf16)f; return p.s[0];     // native cvt
}
__device__ __forceinline__ float bf2f(unsigned short h) {
  PackU p; p.s[0] = h; return (float)p.h[0];
}
__device__ __forceinline__ unsigned packbf(float lo, float hi) {
  PackU p; p.h[0] = (__bf16)lo; p.h[1] = (__bf16)hi; return p.u;
}
__device__ __forceinline__ float frcp(float x) { return __builtin_amdgcn_rcpf(x); }
__device__ __forceinline__ bf16x16 asbf(uintx8 u) { return __builtin_bit_cast(bf16x16, u); }

// A-fragment (16x32 bf16): lane half 'hi' needs elements [hi*16,+8) and [hi*16+8,+8)
__device__ __forceinline__ uintx8 loadA8(const unsigned short* base, int hi) {
  uint4 a = *(const uint4*)(base + hi * 16);
  uint4 b = *(const uint4*)(base + hi * 16 + 8);
  uintx8 r;
  r[0] = a.x; r[1] = a.y; r[2] = a.z; r[3] = a.w;
  r[4] = b.x; r[5] = b.y; r[6] = b.z; r[7] = b.w;
  return r;
}
// B-fragment (32x16 bf16): lane half 'hi' needs elements [hi*16, +16): contiguous 32B
__device__ __forceinline__ uintx8 loadB8(const unsigned short* base, int hi) {
  return *(const uintx8*)(base + hi * 16);
}

// ---------------------------------------------------------------------------
// Phase 1: feature map  fp = sigmoid(norm * X @ P^T) + eps   (bf16 out)
// grid = BH * SEQ/CHK, block = 256 (8 waves); wave w owns row-strip w.
// ---------------------------------------------------------------------------
template <bool WRT>
__global__ __launch_bounds__(256)
void feat_kernel(const float* __restrict__ x, const float* __restrict__ proj,
                 unsigned short* __restrict__ outp,
                 unsigned short* __restrict__ outT) {
  const int bx    = blockIdx.x;
  const int hb    = bx >> 5;
  const int n0    = (bx & 31) * CHK;
  const int lane  = threadIdx.x & 31;
  const int w     = threadIdx.x >> 5;
  const int hi    = lane >> 4;
  const int col16 = lane & 15;
  const long rowBase = (long)hb * SEQ + n0 + w * 16;

  // A fragments: 16 rows x K=64 of normalized x  -> 2 frags (f32 -> bf16)
  uintx8 a[2];
#pragma unroll
  for (int kb = 0; kb < 2; ++kb) {
    const float4* x0 = (const float4*)(x + (rowBase + col16) * DH + kb * 32 + hi * 16);
    const float4* x1 = (const float4*)(x + (rowBase + col16) * DH + kb * 32 + hi * 16 + 8);
#pragma unroll
    for (int ii = 0; ii < 2; ++ii) {
      float4 f0 = x0[ii], f1 = x1[ii];
      a[kb][2 * ii]     = packbf(f0.x * NORMF, f0.y * NORMF);
      a[kb][2 * ii + 1] = packbf(f0.z * NORMF, f0.w * NORMF);
      a[kb][4 + 2 * ii]     = packbf(f1.x * NORMF, f1.y * NORMF);
      a[kb][4 + 2 * ii + 1] = packbf(f1.z * NORMF, f1.w * NORMF);
    }
  }
#pragma unroll 1
  for (int mt = 0; mt < 16; ++mt) {
    floatx8 acc = {};
#pragma unroll
    for (int kb = 0; kb < 2; ++kb) {
      uintx8 bfr;
      const float4* pr = (const float4*)(proj + (mt * 16 + col16) * DH + kb * 32 + hi * 16);
#pragma unroll
      for (int ii = 0; ii < 4; ++ii) {
        float4 f = pr[ii];
        bfr[2 * ii]     = packbf(f.x, f.y);
        bfr[2 * ii + 1] = packbf(f.z, f.w);
      }
      acc = __builtin_amdgcn_wmma_f32_16x16x32_bf16(false, asbf(a[kb]), false, asbf(bfr),
                                                    (short)0, acc, false, false);
    }
#pragma unroll
    for (int j = 0; j < 8; ++j) {
      float s = frcp(1.0f + __expf(-acc[j])) + EPS;
      int  nloc = n0 + w * 16 + j + hi * 8;
      int  m    = mt * 16 + col16;
      unsigned short bs = f2bf(s);
      outp[((long)hb * SEQ + nloc) * NF + m] = bs;
      if (WRT) outT[((long)hb * NF + m) * SEQ + nloc] = bs;
    }
  }
}

// v transposed to [head][e][n] bf16: makes all V B-fragments contiguous.
__global__ __launch_bounds__(256)
void vconv_kernel(const float* __restrict__ v, unsigned short* __restrict__ vt) {
  long i = (long)blockIdx.x * blockDim.x + threadIdx.x;   // over BH*DH*SEQ
  long n  = i & (SEQ - 1);
  long r  = i >> 12;            // hb*DH + e
  long hb = r >> 6;
  long e  = r & (DH - 1);
  vt[i] = f2bf(v[(hb * SEQ + n) * DH + e]);
}

// ---------------------------------------------------------------------------
// Phase 2: per-chunk summaries:  ckv = K_c^T @ V_c  [256x64] f32,
//          cks = per-chunk feature sums of kp  [256] f32.
// grid = BH*NCHK, block = 256; wave w owns m-tiles {2w, 2w+1}.
// ---------------------------------------------------------------------------
__global__ __launch_bounds__(256)
void chunksum_kernel(const unsigned short* __restrict__ kpT,
                     const unsigned short* __restrict__ vt,
                     float* __restrict__ ckv, float* __restrict__ cks) {
  const int bx    = blockIdx.x;
  const int hb    = bx >> 5;
  const int c     = bx & 31;
  const int n0    = c * CHK;
  const int lane  = threadIdx.x & 31;
  const int w     = threadIdx.x >> 5;
  const int hi    = lane >> 4;
  const int col16 = lane & 15;

  const long obase = (long)(hb * NCHK + c) * (NF * DH);
#pragma unroll 1
  for (int mi = 0; mi < 2; ++mi) {
    int f = (w * 2 + mi) * 16 + col16;
    const unsigned short* kr = kpT + ((long)hb * NF + f) * SEQ + n0;
    uintx8 a[4];
#pragma unroll
    for (int kb = 0; kb < 4; ++kb) a[kb] = loadA8(kr + kb * 32, hi);
#pragma unroll 1
    for (int et = 0; et < 4; ++et) {
      floatx8 acc = {};
      int e = et * 16 + col16;
      const unsigned short* vr = vt + ((long)hb * DH + e) * SEQ + n0;
#pragma unroll
      for (int kb = 0; kb < 4; ++kb) {
        acc = __builtin_amdgcn_wmma_f32_16x16x32_bf16(false, asbf(a[kb]), false,
                                                      asbf(loadB8(vr + kb * 32, hi)),
                                                      (short)0, acc, false, false);
      }
#pragma unroll
      for (int j = 0; j < 8; ++j) {
        int m = (w * 2 + mi) * 16 + j + hi * 8;
        ckv[obase + (long)m * DH + e] = acc[j];
      }
    }
  }
  // per-chunk feature sums (contiguous along n in kpT, b128 reads)
  {
    int f = threadIdx.x;
    const uint4* kr = (const uint4*)(kpT + ((long)hb * NF + f) * SEQ + n0);
    float s = 0.f;
#pragma unroll 4
    for (int b = 0; b < CHK / 8; ++b) {
      uint4 u = kr[b];
      s += bf2f((unsigned short)u.x) + bf2f((unsigned short)(u.x >> 16))
         + bf2f((unsigned short)u.y) + bf2f((unsigned short)(u.y >> 16))
         + bf2f((unsigned short)u.z) + bf2f((unsigned short)(u.z >> 16))
         + bf2f((unsigned short)u.w) + bf2f((unsigned short)(u.w >> 16));
    }
    cks[(long)(hb * NCHK + c) * NF + f] = s;
  }
}

// ---------------------------------------------------------------------------
// Phase 3: exclusive prefix over chunks (register-resident accumulators).
// Thread t owns e = t&63, m in [(t>>6)*64, +64): packed b128 stores of bf16.
// ---------------------------------------------------------------------------
__global__ __launch_bounds__(256)
void prefix_kernel(const float* __restrict__ ckv, const float* __restrict__ cks,
                   unsigned short* __restrict__ sprevT, float* __restrict__ ksprev) {
  const int hb = blockIdx.x;
  const int t  = threadIdx.x;
  const int e  = t & 63;
  const int m0 = (t >> 6) * 64;
  float acc[64];
#pragma unroll
  for (int i = 0; i < 64; ++i) acc[i] = 0.f;
  for (int c = 0; c < NCHK; ++c) {
    long base = (long)(hb * NCHK + c) * (NF * DH);
    unsigned short* sp = sprevT + base + (long)e * NF + m0;
#pragma unroll
    for (int i = 0; i < 64; i += 8) {
      uint4 pk;
      pk.x = packbf(acc[i + 0], acc[i + 1]);
      pk.y = packbf(acc[i + 2], acc[i + 3]);
      pk.z = packbf(acc[i + 4], acc[i + 5]);
      pk.w = packbf(acc[i + 6], acc[i + 7]);
      *(uint4*)(sp + i) = pk;            // exclusive prefix, bf16
#pragma unroll
      for (int u = 0; u < 8; ++u)
        acc[i + u] += ckv[base + (long)(m0 + i + u) * DH + e];
    }
  }
  float s = 0.f;
  for (int c = 0; c < NCHK; ++c) {
    long idx = (long)(hb * NCHK + c) * NF + t;
    ksprev[idx] = s;
    s += cks[idx];
  }
}

// ---------------------------------------------------------------------------
// Phase 4: per-chunk output:
//   A = tril(Q_c K_c^T), out = (Q_c @ S_prev + A @ V_c) * Dinv
// grid = BH*NCHK, block = 256; wave w owns row-strip w.
// ---------------------------------------------------------------------------
__global__ __launch_bounds__(256)
void out_kernel(const unsigned short* __restrict__ qp,
                const unsigned short* __restrict__ kp,
                const unsigned short* __restrict__ vt,
                const unsigned short* __restrict__ sprevT,
                const float* __restrict__ ksprev,
                float* __restrict__ out) {
  __shared__ unsigned short s_aqk[CHK * CHK];   // 32 KB masked scores (bf16)
  __shared__ float s_ksum[NF];
  __shared__ float s_dinv[CHK];

  const int bx    = blockIdx.x;
  const int hb    = bx >> 5;
  const int c     = bx & 31;
  const int n0    = c * CHK;
  const long gb   = (long)hb * SEQ + n0;
  const int lane  = threadIdx.x & 31;
  const int w     = threadIdx.x >> 5;
  const int hi    = lane >> 4;
  const int col16 = lane & 15;

  s_ksum[threadIdx.x] = ksprev[(long)(hb * NCHK + c) * NF + threadIdx.x];

  // qp A fragments: strip rows, K = 256 features -> 8 frags (reused for QK^T and Q @ S_prev)
  uintx8 aq[8];
  {
    const unsigned short* qr = qp + (gb + w * 16 + col16) * NF;
#pragma unroll
    for (int kb = 0; kb < 8; ++kb) aq[kb] = loadA8(qr + kb * 32, hi);
  }

  // causal Q K^T
#pragma unroll 1
  for (int ct = 0; ct < 8; ++ct) {
    if (ct > w) {   // wave-uniform: fully masked tile -> zero fill, no WMMA
#pragma unroll
      for (int j = 0; j < 8; ++j)
        s_aqk[(w * 16 + j + hi * 8) * CHK + ct * 16 + col16] = 0;
      continue;
    }
    floatx8 acc = {};
    const unsigned short* kr = kp + (gb + ct * 16 + col16) * NF;
#pragma unroll
    for (int kb = 0; kb < 8; ++kb) {
      acc = __builtin_amdgcn_wmma_f32_16x16x32_bf16(false, asbf(aq[kb]), false,
                                                    asbf(loadB8(kr + kb * 32, hi)),
                                                    (short)0, acc, false, false);
    }
#pragma unroll
    for (int j = 0; j < 8; ++j) {
      int rloc = j + hi * 8;
      float vval = acc[j];
      if (ct == w && col16 > rloc) vval = 0.f;   // strict upper triangle masked
      s_aqk[(w * 16 + rloc) * CHK + ct * 16 + col16] = f2bf(vval);
    }
  }
  __syncthreads();

  // denominators: q.k_cum = q.ksum_prev + rowsum(tril A);  + eps * sum(q)
  if (threadIdx.x < CHK) {
    int row = threadIdx.x;
    float rs = 0.f;
    const uint4* ar = (const uint4*)(s_aqk + row * CHK);
#pragma unroll 4
    for (int b = 0; b < CHK / 8; ++b) {
      uint4 u = ar[b];
      rs += bf2f((unsigned short)u.x) + bf2f((unsigned short)(u.x >> 16))
          + bf2f((unsigned short)u.y) + bf2f((unsigned short)(u.y >> 16))
          + bf2f((unsigned short)u.z) + bf2f((unsigned short)(u.z >> 16))
          + bf2f((unsigned short)u.w) + bf2f((unsigned short)(u.w >> 16));
    }
    const uint4* qr = (const uint4*)(qp + (gb + row) * NF);
    float qd = 0.f, qs = 0.f;
#pragma unroll 4
    for (int b = 0; b < NF / 8; ++b) {
      uint4 u = qr[b];
      float q0 = bf2f((unsigned short)u.x), q1 = bf2f((unsigned short)(u.x >> 16));
      float q2 = bf2f((unsigned short)u.y), q3 = bf2f((unsigned short)(u.y >> 16));
      float q4 = bf2f((unsigned short)u.z), q5 = bf2f((unsigned short)(u.z >> 16));
      float q6 = bf2f((unsigned short)u.w), q7 = bf2f((unsigned short)(u.w >> 16));
      const float* ks = s_ksum + 8 * b;
      qd += q0 * ks[0] + q1 * ks[1] + q2 * ks[2] + q3 * ks[3]
          + q4 * ks[4] + q5 * ks[5] + q6 * ks[6] + q7 * ks[7];
      qs += q0 + q1 + q2 + q3 + q4 + q5 + q6 + q7;
    }
    s_dinv[row] = frcp(qd + rs + EPS * qs);
  }
  __syncthreads();

  // A fragments of masked scores from LDS (K = 128 -> 4 frags, ds_load_b128)
  uintx8 aa[4];
  {
    const unsigned short* ar = s_aqk + (w * 16 + col16) * CHK;
#pragma unroll
    for (int kb = 0; kb < 4; ++kb) aa[kb] = loadA8(ar + kb * 32, hi);
  }

  const long sbase = (long)(hb * NCHK + c) * (NF * DH);
#pragma unroll 1
  for (int et = 0; et < 4; ++et) {
    floatx8 acc = {};
    int e = et * 16 + col16;
    // O_inter = Q @ S_prev   (B from transposed S_prev: contiguous b128)
    const unsigned short* sr = sprevT + sbase + (long)e * NF;
#pragma unroll
    for (int kb = 0; kb < 8; ++kb) {
      acc = __builtin_amdgcn_wmma_f32_16x16x32_bf16(false, asbf(aq[kb]), false,
                                                    asbf(loadB8(sr + kb * 32, hi)),
                                                    (short)0, acc, false, false);
    }
    // O_intra = tril(A) @ V  (B from transposed V: contiguous b128)
    const unsigned short* vr = vt + ((long)hb * DH + e) * SEQ + n0;
#pragma unroll
    for (int kb = 0; kb < 4; ++kb) {
      acc = __builtin_amdgcn_wmma_f32_16x16x32_bf16(false, asbf(aa[kb]), false,
                                                    asbf(loadB8(vr + kb * 32, hi)),
                                                    (short)0, acc, false, false);
    }
#pragma unroll
    for (int j = 0; j < 8; ++j) {
      int row = w * 16 + j + hi * 8;
      out[(gb + row) * DH + e] = acc[j] * s_dinv[row];
    }
  }
}

// ---------------------------------------------------------------------------
extern "C" void kernel_launch(void* const* d_in, const int* in_sizes, int n_in,
                              void* d_out, int out_size, void* d_ws, size_t ws_size,
                              hipStream_t stream) {
  const float* q    = (const float*)d_in[0];
  const float* k    = (const float*)d_in[1];
  const float* v    = (const float*)d_in[2];
  const float* proj = (const float*)d_in[3];
  float* out = (float*)d_out;

  char* ws = (char*)d_ws;
  size_t off = 0;
  unsigned short* qp  = (unsigned short*)(ws + off); off += (size_t)BH * SEQ * NF * 2;
  unsigned short* kp  = (unsigned short*)(ws + off); off += (size_t)BH * SEQ * NF * 2;
  unsigned short* kpT = (unsigned short*)(ws + off); off += (size_t)BH * NF * SEQ * 2;
  unsigned short* vt  = (unsigned short*)(ws + off); off += (size_t)BH * DH * SEQ * 2;
  float* ckv = (float*)(ws + off);                   off += (size_t)BH * NCHK * NF * DH * 4;
  float* cks = (float*)(ws + off);                   off += (size_t)BH * NCHK * NF * 4;
  unsigned short* sprevT = (unsigned short*)(ws + off); off += (size_t)BH * NCHK * NF * DH * 2;
  float* ksprev = (float*)(ws + off);                off += (size_t)BH * NCHK * NF * 4;

  dim3 blk(256);
  feat_kernel<false><<<BH * (SEQ / CHK), blk, 0, stream>>>(q, proj, qp, (unsigned short*)0);
  feat_kernel<true><<<BH * (SEQ / CHK), blk, 0, stream>>>(k, proj, kp, kpT);
  long nv = (long)BH * SEQ * DH;
  vconv_kernel<<<(int)(nv / 256), blk, 0, stream>>>(v, vt);
  chunksum_kernel<<<BH * NCHK, blk, 0, stream>>>(kpT, vt, ckv, cks);
  prefix_kernel<<<BH, blk, 0, stream>>>(ckv, cks, sprevT, ksprev);
  out_kernel<<<BH * NCHK, blk, 0, stream>>>(qp, kp, vt, sprevT, ksprev, out);
}
